// PhysicsInformedDiffusion_62466004353659
// MI455X (gfx1250) — compile-verified, matrix operand order checked by reading, not run on hardware
//
#include <hip/hip_runtime.h>
#include <math.h>

typedef _Float16 half_t;
typedef __attribute__((ext_vector_type(16))) _Float16 v16h;
typedef __attribute__((ext_vector_type(8)))  float    v8f;

union FragH { v16h v; uint4 q[2]; };

__device__ __forceinline__ FragH load_frag(const half_t* p)
{
    FragH f;
    const uint4* q = (const uint4*)p;
    f.q[0] = q[0];           // K = k0   .. k0+7
    f.q[1] = q[2];           // K = k0+16.. k0+23
    return f;
}

__device__ __forceinline__ FragH load_frag_pred(const half_t* p, bool inb)
{
    FragH f;
    if (inb) {
        const uint4* q = (const uint4*)p;
        f.q[0] = q[0];
        f.q[1] = q[2];
    } else {
        f.q[0] = make_uint4(0u, 0u, 0u, 0u);
        f.q[1] = make_uint4(0u, 0u, 0u, 0u);
    }
    return f;
}

// ---------------------------------------------------------------------------
// Implicit-GEMM 3x3 SAME conv using v_wmma_f32_16x16x32_f16.
// X:  [B,H,W,Cin]  f16 channel-last (Cin multiple of 32)
// W9: [9][Cout][Cin] f16 (tap-major, cross-correlation taps kh*3+kw)
// Y:  [B*H*W, Cout] f32 channel-last
// One wave computes a 16(Cout) x 64(pixel) tile as 4 sub-tiles sharing the
// A (weight) fragment -> 4x A-reuse, 4 independent accumulators.
// ---------------------------------------------------------------------------
__global__ __launch_bounds__(128)
void conv3x3_wmma_kernel(const half_t* __restrict__ X,
                         const half_t* __restrict__ W9,
                         const float*  __restrict__ bias,
                         float* __restrict__ Y,
                         int H, int W, int Cin, int Cout)
{
    const int lane  = threadIdx.x & 31;
    const int wv    = threadIdx.x >> 5;
    const int HW    = H * W;
    const int base  = (blockIdx.x * 4 + wv) * 64;   // first pixel of this wave
    const int mbase = blockIdx.y * 16;
    const int ln16  = lane & 15;
    const int g     = lane >> 4;                    // half-wave K-split group

    int pixs[4], hv[4], wpv[4], bv[4];
#pragma unroll
    for (int st = 0; st < 4; ++st) {
        int pix = base + st * 16 + ln16;
        pixs[st] = pix;
        int b = pix / HW;
        int hw = pix - b * HW;
        bv[st] = b;
        hv[st] = hw / W;
        wpv[st] = hw - hv[st] * W;
    }

    v8f acc[4] = {{}, {}, {}, {}};

    // A row for this lane: M = mbase + ln16, K chunk starts at 8*g
    const long long arow = (long long)(mbase + ln16) * Cin + 8 * g;

    for (int tap = 0; tap < 9; ++tap) {
        const int kh = tap / 3;
        const int dh = kh - 1;
        const int dw = (tap - kh * 3) - 1;
        const half_t* wp = W9 + (long long)tap * Cout * Cin + arow;

        const half_t* xp[4];
        bool inb[4];
#pragma unroll
        for (int st = 0; st < 4; ++st) {
            int hs = hv[st] + dh;
            int ws = wpv[st] + dw;
            inb[st] = (hs >= 0) & (hs < H) & (ws >= 0) & (ws < W);
            xp[st] = X + ((long long)(bv[st] * HW + hs * W + ws) * Cin + 8 * g);
        }

        for (int kb = 0; kb < Cin; kb += 32) {
            FragH a = load_frag(wp + kb);
            __builtin_prefetch((const void*)(wp + kb + 32), 0, 0);
            FragH b0 = load_frag_pred(xp[0] + kb, inb[0]);
            FragH b1 = load_frag_pred(xp[1] + kb, inb[1]);
            FragH b2 = load_frag_pred(xp[2] + kb, inb[2]);
            FragH b3 = load_frag_pred(xp[3] + kb, inb[3]);
            acc[0] = __builtin_amdgcn_wmma_f32_16x16x32_f16(
                         false, a.v, false, b0.v, (short)0, acc[0], false, false);
            acc[1] = __builtin_amdgcn_wmma_f32_16x16x32_f16(
                         false, a.v, false, b1.v, (short)0, acc[1], false, false);
            acc[2] = __builtin_amdgcn_wmma_f32_16x16x32_f16(
                         false, a.v, false, b2.v, (short)0, acc[2], false, false);
            acc[3] = __builtin_amdgcn_wmma_f32_16x16x32_f16(
                         false, a.v, false, b3.v, (short)0, acc[3], false, false);
        }
    }

    // D layout: lane stores M = mbase + 8*g + j (j=0..7) at its pixel
    const float* bp = bias + mbase + 8 * g;
#pragma unroll
    for (int st = 0; st < 4; ++st) {
        float* yp = Y + (long long)pixs[st] * Cout + mbase + 8 * g;
#pragma unroll
        for (int j = 0; j < 8; ++j) yp[j] = acc[st][j] + bp[j];
    }
}

// ---------------------------------------------------------------------------
// Weight repack: [Cout,Cin,3,3] f32 -> [tap][Cout][Cinp] f16 (zero padded)
// ---------------------------------------------------------------------------
__global__ void convert_w_kernel(const float* __restrict__ w, half_t* __restrict__ dst,
                                 int Cout, int Cin, int Cinp, int total)
{
    int i = blockIdx.x * blockDim.x + threadIdx.x;
    if (i >= total) return;
    int ci  = i % Cinp;
    int r   = i / Cinp;
    int co  = r % Cout;
    int tap = r / Cout;
    float v = 0.f;
    if (ci < Cin) v = w[((long long)(co * Cin) + ci) * 9 + tap];
    dst[i] = (half_t)v;
}

// x_t [B,2,H,W] f32 -> [B,H,W,32] f16 (channels 2..31 zero)
__global__ void pack_x_kernel(const float* __restrict__ x, half_t* __restrict__ dst, int total)
{
    int i = blockIdx.x * blockDim.x + threadIdx.x;
    if (i >= total) return;
    int c = i & 31;
    int p = i >> 5;               // global pixel 0..B*HW-1
    float v = 0.f;
    if (c < 2) {
        int b = p >> 16;          // HW = 65536
        int n = p & 65535;
        v = x[(long long)(b * 2 + c) * 65536 + n];
    }
    dst[i] = (half_t)v;
}

// GroupNorm stats: one block per (b,group); mv[bg] = {mean, var}
__global__ void gn_stats_kernel(const float* __restrict__ Y, float* __restrict__ mv,
                                int HW, int C)
{
    int bg = blockIdx.x;
    int b = bg >> 3, g = bg & 7;
    int cg = C >> 3;
    int n  = HW * cg;
    float s1 = 0.f, s2 = 0.f;
    for (int i = threadIdx.x; i < n; i += blockDim.x) {
        int p = i / cg, c = i - p * cg;
        float v = Y[(long long)(b * HW + p) * C + g * cg + c];
        s1 += v; s2 += v * v;
    }
    __shared__ float sh1[256], sh2[256];
    sh1[threadIdx.x] = s1; sh2[threadIdx.x] = s2;
    __syncthreads();
    for (int s = 128; s > 0; s >>= 1) {
        if ((int)threadIdx.x < s) {
            sh1[threadIdx.x] += sh1[threadIdx.x + s];
            sh2[threadIdx.x] += sh2[threadIdx.x + s];
        }
        __syncthreads();
    }
    if (threadIdx.x == 0) {
        float m = sh1[0] / (float)n;
        mv[bg * 2]     = m;
        mv[bg * 2 + 1] = sh2[0] / (float)n - m * m;
    }
}

// Normalize + affine + SiLU + f16 convert, write at channel offset (concat)
__global__ void gn_act_kernel(const float* __restrict__ Y, const float* __restrict__ mv,
                              const float* __restrict__ gamma, const float* __restrict__ beta,
                              half_t* __restrict__ dst, int HW, int C, int Cdst, int coff, int total)
{
    int i = blockIdx.x * blockDim.x + threadIdx.x;
    if (i >= total) return;
    int c = i % C;
    int p = i / C;
    int b = p / HW;
    int g = c / (C >> 3);
    float m  = mv[(b * 8 + g) * 2];
    float vr = mv[(b * 8 + g) * 2 + 1];
    float v = (Y[i] - m) * rsqrtf(vr + 1e-5f) * gamma[c] + beta[c];
    v = v / (1.f + expf(-v));     // SiLU
    dst[(long long)p * Cdst + coff + c] = (half_t)v;
}

// 2x2 max pool, channel-last f16
__global__ void pool_kernel(const half_t* __restrict__ src, half_t* __restrict__ dst,
                            int Hs, int Ws, int C, int total)
{
    int i = blockIdx.x * blockDim.x + threadIdx.x;
    if (i >= total) return;
    int c = i % C; int r = i / C;
    int Wo = Ws >> 1, Ho = Hs >> 1;
    int wo = r % Wo; r /= Wo;
    int ho = r % Ho; int b = r / Ho;
    const half_t* s = src + ((long long)((b * Hs + 2 * ho) * Ws + 2 * wo)) * C + c;
    long long rs = (long long)Ws * C;
    float v0 = (float)s[0], v1 = (float)s[C];
    float v2 = (float)s[rs], v3 = (float)s[rs + C];
    dst[i] = (half_t)fmaxf(fmaxf(v0, v1), fmaxf(v2, v3));
}

// Bilinear 2x upsample (half-pixel centers) + optional temb add; writes at coff=0
__global__ void up2_kernel(const half_t* __restrict__ src, half_t* __restrict__ dst,
                           const float* __restrict__ temb,
                           int Hs, int Ws, int Cs, int Cdst, int total)
{
    int i = blockIdx.x * blockDim.x + threadIdx.x;
    if (i >= total) return;
    int c = i % Cs; int r = i / Cs;
    int Wo = Ws * 2, Ho = Hs * 2;
    int wo = r % Wo; r /= Wo;
    int ho = r % Ho; int b = r / Ho;
    float fy = 0.5f * ho - 0.25f;
    int y0 = (int)floorf(fy);
    float wy = fy - (float)y0;
    int y1 = min(y0 + 1, Hs - 1); y0 = max(y0, 0);
    float fx = 0.5f * wo - 0.25f;
    int x0 = (int)floorf(fx);
    float wx = fx - (float)x0;
    int x1 = min(x0 + 1, Ws - 1); x0 = max(x0, 0);
    const half_t* sb = src + (long long)b * Hs * Ws * Cs;
    float s00 = (float)sb[((long long)(y0 * Ws) + x0) * Cs + c];
    float s01 = (float)sb[((long long)(y0 * Ws) + x1) * Cs + c];
    float s10 = (float)sb[((long long)(y1 * Ws) + x0) * Cs + c];
    float s11 = (float)sb[((long long)(y1 * Ws) + x1) * Cs + c];
    float v = (1.f - wy) * ((1.f - wx) * s00 + wx * s01)
            +        wy  * ((1.f - wx) * s10 + wx * s11);
    if (temb) v += temb[b * 256 + c];
    dst[((long long)(b * Ho + ho) * Wo + wo) * Cdst + c] = (half_t)v;
}

__global__ void concat_copy_kernel(const half_t* __restrict__ src, half_t* __restrict__ dst,
                                   int Csrc, int Cdst, int coff, int total)
{
    int i = blockIdx.x * blockDim.x + threadIdx.x;
    if (i >= total) return;
    int c = i % Csrc; int p = i / Csrc;
    dst[(long long)p * Cdst + coff + c] = src[i];
}

// timestep embedding + 2-layer MLP (B=4, 128 -> 256 -> 256), one block
__global__ void temb_kernel(const int* __restrict__ t,
                            const float* __restrict__ w1, const float* __restrict__ b1,
                            const float* __restrict__ w2, const float* __restrict__ b2,
                            float* __restrict__ temb)
{
    __shared__ float emb[4][128];
    __shared__ float hid[4][256];
    int tid = threadIdx.x;
    for (int i = tid; i < 4 * 128; i += 256) {
        int b = i / 128, k = i % 128;
        int hk = k & 63;
        float freq = expf((float)hk * (-logf(10000.f) / 63.f));
        float e = (float)t[b] * freq;
        emb[b][k] = (k >= 64) ? cosf(e) : sinf(e);
    }
    __syncthreads();
    for (int i = tid; i < 4 * 256; i += 256) {
        int b = i / 256, j = i % 256;
        float s = b1[j];
        for (int k = 0; k < 128; ++k) s += emb[b][k] * w1[k * 256 + j];
        hid[b][j] = s / (1.f + expf(-s));
    }
    __syncthreads();
    for (int i = tid; i < 4 * 256; i += 256) {
        int b = i / 256, j = i % 256;
        float s = b2[j];
        for (int k = 0; k < 256; ++k) s += hid[b][k] * w2[k * 256 + j];
        temb[b * 256 + j] = s;
    }
}

// 1x1 output conv: [B,H,W,64] f16 -> noise_pred [B,2,HW] f32
__global__ void out_conv_kernel(const half_t* __restrict__ X, const float* __restrict__ w,
                                const float* __restrict__ bias, float* __restrict__ np_,
                                int HW, int total)
{
    int i = blockIdx.x * blockDim.x + threadIdx.x;
    if (i >= total) return;
    int c = i & 1;
    int p = i >> 1;
    int b = p / HW, n = p - b * HW;
    float s = bias[c];
    const half_t* xp = X + (long long)p * 64;
#pragma unroll
    for (int k = 0; k < 64; ++k) s += (float)xp[k] * w[c * 64 + k];
    np_[(long long)(b * 2 + c) * HW + n] = s;
}

// per-batch diffusion coefficients: ab, ab_prev (neg-index wrap), beta, t0-flag
__global__ void coef_kernel(const int* __restrict__ t, float* __restrict__ cf)
{
    int b = threadIdx.x;
    if (b >= 4) return;
    int tb = t[b];
    int ip = tb - 1; if (ip < 0) ip += 1000;   // jnp wraparound indexing
    float ab = 1.f, abp = 1.f;
    for (int i = 0; i < 1000; ++i) {
        float beta = 1e-4f + (0.02f - 1e-4f) * (float)i / 999.f;
        float a = 1.f - beta;
        if (i <= tb) ab  *= a;
        if (i <= ip) abp *= a;
    }
    cf[b * 4 + 0] = ab;
    cf[b * 4 + 1] = abp;
    cf[b * 4 + 2] = 1e-4f + (0.02f - 1e-4f) * (float)tb / 999.f;
    cf[b * 4 + 3] = (t[0] > 0) ? 1.f : 0.f;
}

__global__ void x0_kernel(const float* __restrict__ xt, const float* __restrict__ np_,
                          const float* __restrict__ cf, float* __restrict__ vx,
                          float* __restrict__ vy, int HW, int total)
{
    int i = blockIdx.x * blockDim.x + threadIdx.x;
    if (i >= total) return;
    int b = i / HW, n = i - b * HW;
    float ab = cf[b * 4];
    float s1 = sqrtf(1.f - ab);
    float inv = 1.f / sqrtf(ab);
    vx[i] = (xt[(long long)(b * 2) * HW + n]     - s1 * np_[(long long)(b * 2) * HW + n])     * inv;
    vy[i] = (xt[(long long)(b * 2 + 1) * HW + n] - s1 * np_[(long long)(b * 2 + 1) * HW + n]) * inv;
}

__global__ void div_kernel(const float* __restrict__ vx, const float* __restrict__ vy,
                           float* __restrict__ dv, int H, int W, int total)
{
    int i = blockIdx.x * blockDim.x + threadIdx.x;
    if (i >= total) return;
    int w = i % W; int r = i / W; int h = r % H; int b = r / H;
    const float* vxb = vx + (long long)b * H * W;
    const float* vyb = vy + (long long)b * H * W;
    int wc = min(max(w, 1), W - 2);
    int hc = min(max(h, 1), H - 2);
    float dvx = 0.5f * (vxb[h * W + wc + 1] - vxb[h * W + wc - 1]);
    float dvy = 0.5f * (vyb[(hc + 1) * W + w] - vyb[(hc - 1) * W + w]);
    dv[i] = dvx + dvy;
}

__global__ void jacobi_init_kernel(const float* __restrict__ dv, float* __restrict__ out, int total)
{
    int i = blockIdx.x * blockDim.x + threadIdx.x;
    if (i >= total) return;
    out[i] = -0.25f * dv[i];
}

__global__ void jacobi_kernel(const float* __restrict__ phi, const float* __restrict__ dv,
                              float* __restrict__ out, int H, int W, int total)
{
    int i = blockIdx.x * blockDim.x + threadIdx.x;
    if (i >= total) return;
    int w = i % W; int r = i / W; int h = r % H; int b = r / H;
    const float* p = phi + (long long)b * H * W;
    float l = p[h * W + max(w - 1, 0)];
    float rr = p[h * W + min(w + 1, W - 1)];
    float u = p[max(h - 1, 0) * W + w];
    float d = p[min(h + 1, H - 1) * W + w];
    out[i] = 0.25f * (l + rr + u + d - dv[i]);
}

// projection + posterior sample; writes x_prev then x0 into d_out
__global__ void final_kernel(const float* __restrict__ phi, const float* __restrict__ vx,
                             const float* __restrict__ vy, const float* __restrict__ np_,
                             const float* __restrict__ noise, const float* __restrict__ cf,
                             float* __restrict__ out, int H, int W, int total)
{
    int i = blockIdx.x * blockDim.x + threadIdx.x;
    if (i >= total) return;
    int w = i % W; int r = i / W; int h = r % H; int b = r / H;
    int HW = H * W;
    int n = h * W + w;
    const float* p = phi + (long long)b * HW;
    float l = p[h * W + max(w - 1, 0)];
    float rr = p[h * W + min(w + 1, W - 1)];
    float u = p[max(h - 1, 0) * W + w];
    float d = p[min(h + 1, H - 1) * W + w];
    float x0x = vx[i] - (rr - l);
    float x0y = vy[i] - (d - u);
    float ab = cf[b * 4], abp = cf[b * 4 + 1], beta = cf[b * 4 + 2], flag = cf[b * 4 + 3];
    float pv = beta * (1.f - abp) / (1.f - ab);
    float cA = sqrtf(abp);
    float cB = sqrtf(fmaxf(1.f - abp - pv, 0.f));
    float cC = sqrtf(pv);
    float npx = np_[(long long)(b * 2) * HW + n];
    float npy = np_[(long long)(b * 2 + 1) * HW + n];
    float nzx = noise[(long long)(b * 2) * HW + n];
    float nzy = noise[(long long)(b * 2 + 1) * HW + n];
    float xpx = cA * x0x + cB * npx + cC * nzx;
    float xpy = cA * x0y + cB * npy + cC * nzy;
    long long X0OFF = (long long)4 * 2 * HW;
    out[(long long)(b * 2) * HW + n]             = (flag > 0.5f) ? xpx : x0x;
    out[(long long)(b * 2 + 1) * HW + n]         = (flag > 0.5f) ? xpy : x0y;
    out[X0OFF + (long long)(b * 2) * HW + n]     = x0x;
    out[X0OFF + (long long)(b * 2 + 1) * HW + n] = x0y;
}

// ===========================================================================
extern "C" void kernel_launch(void* const* d_in, const int* in_sizes, int n_in,
                              void* d_out, int out_size, void* d_ws, size_t ws_size,
                              hipStream_t stream)
{
    (void)in_sizes; (void)n_in; (void)out_size; (void)ws_size;

    const float* x_t   = (const float*)d_in[0];
    const int*   tarr  = (const int*)d_in[1];
    const float* noise = (const float*)d_in[2];
    const float* te_w1 = (const float*)d_in[3];
    const float* te_b1 = (const float*)d_in[4];
    const float* te_w2 = (const float*)d_in[5];
    const float* te_b2 = (const float*)d_in[6];
    float* out = (float*)d_out;

    // bump allocator over d_ws
    char* wsP = (char*)d_ws;
    auto alloc = [&](size_t bytes) -> void* {
        void* r = (void*)wsP;
        wsP += (bytes + 255) & ~(size_t)255;
        return r;
    };

    // conv layer table: {param index of w, Cout, Cin, Cin padded}
    struct CL { int idx, Cout, Cin, Cinp; };
    const CL convs[14] = {
        { 7,  64,   2,  32}, {11,  64,  64,  64},
        {15, 128,  64,  64}, {19, 128, 128, 128},
        {23, 256, 128, 128}, {27, 256, 256, 256},
        {31, 256, 256, 256}, {35, 256, 256, 256},
        {39, 128, 512, 512}, {43, 128, 128, 128},
        {47,  64, 256, 256}, {51,  64,  64,  64},
        {55,  64, 128, 128}, {59,  64,  64,  64},
    };

    half_t* Wh[14];
    for (int l = 0; l < 14; ++l) {
        size_t ne = (size_t)9 * convs[l].Cout * convs[l].Cinp;
        Wh[l] = (half_t*)alloc(ne * sizeof(half_t));
        int th = (int)ne;
        convert_w_kernel<<<(th + 255) / 256, 256, 0, stream>>>(
            (const float*)d_in[convs[l].idx], Wh[l],
            convs[l].Cout, convs[l].Cin, convs[l].Cinp, th);
    }

    // activation buffers (f16 channel-last)
    half_t* xin   = (half_t*)alloc((size_t)4 * 65536 * 32 * 2);
    half_t* e1b   = (half_t*)alloc((size_t)4 * 65536 * 64 * 2);
    half_t* p1b   = (half_t*)alloc((size_t)4 * 16384 * 64 * 2);
    half_t* e2b   = (half_t*)alloc((size_t)4 * 16384 * 128 * 2);
    half_t* p2b   = (half_t*)alloc((size_t)4 * 4096 * 128 * 2);
    half_t* e3b   = (half_t*)alloc((size_t)4 * 4096 * 256 * 2);
    half_t* p3b   = (half_t*)alloc((size_t)4 * 1024 * 256 * 2);
    half_t* botb  = (half_t*)alloc((size_t)4 * 1024 * 256 * 2);
    half_t* d3in  = (half_t*)alloc((size_t)4 * 4096 * 512 * 2);
    half_t* d3out = (half_t*)alloc((size_t)4 * 4096 * 128 * 2);
    half_t* d2in  = (half_t*)alloc((size_t)4 * 16384 * 256 * 2);
    half_t* d2out = (half_t*)alloc((size_t)4 * 16384 * 64 * 2);
    half_t* d1in  = (half_t*)alloc((size_t)4 * 65536 * 128 * 2);
    half_t* d1out = (half_t*)alloc((size_t)4 * 65536 * 64 * 2);
    float*  convf = (float*)alloc((size_t)4 * 65536 * 64 * 4);   // conv f32 scratch (max)
    half_t* midb  = (half_t*)alloc((size_t)4 * 65536 * 64 * 2);  // convblock mid (max)
    float*  mv    = (float*)alloc(64 * sizeof(float));
    float*  temb  = (float*)alloc(4 * 256 * sizeof(float));
    float*  npred = (float*)alloc((size_t)4 * 2 * 65536 * 4);
    float*  cf    = (float*)alloc(16 * sizeof(float));
    float*  vx    = (float*)alloc((size_t)4 * 65536 * 4);
    float*  vy    = (float*)alloc((size_t)4 * 65536 * 4);
    float*  dvb   = (float*)alloc((size_t)4 * 65536 * 4);
    float*  phiA  = (float*)alloc((size_t)4 * 65536 * 4);
    float*  phiB  = (float*)alloc((size_t)4 * 65536 * 4);

    // input pack + timestep MLP
    {
        int th = 4 * 65536 * 32;
        pack_x_kernel<<<(th + 255) / 256, 256, 0, stream>>>(x_t, xin, th);
    }
    temb_kernel<<<1, 256, 0, stream>>>(tarr, te_w1, te_b1, te_w2, te_b2, temb);

    auto run_conv = [&](const half_t* xsrc, const half_t* wh, const float* bias,
                        int H, int W, int Cinp, int Cout, float* yout) {
        dim3 grid((4 * H * W) / 256, Cout / 16);   // 4 waves/block, 64 px/wave
        conv3x3_wmma_kernel<<<grid, 128, 0, stream>>>(xsrc, wh, bias, yout, H, W, Cinp, Cout);
    };
    auto run_gn = [&](const float* y, const float* gmm, const float* bet,
                      half_t* dst, int H, int W, int C, int Cdst, int coff) {
        gn_stats_kernel<<<32, 256, 0, stream>>>(y, mv, H * W, C);
        int tot = 4 * H * W * C;
        gn_act_kernel<<<(tot + 255) / 256, 256, 0, stream>>>(y, mv, gmm, bet, dst,
                                                             H * W, C, Cdst, coff, tot);
    };
    auto run_block = [&](const half_t* xsrc, int li, int pidx, int H, int W,
                         int Cout, half_t* dst) {
        run_conv(xsrc, Wh[li], (const float*)d_in[pidx + 1], H, W, convs[li].Cinp, Cout, convf);
        run_gn(convf, (const float*)d_in[pidx + 2], (const float*)d_in[pidx + 3],
               midb, H, W, Cout, Cout, 0);
        run_conv(midb, Wh[li + 1], (const float*)d_in[pidx + 5], H, W, convs[li + 1].Cinp, Cout, convf);
        run_gn(convf, (const float*)d_in[pidx + 6], (const float*)d_in[pidx + 7],
               dst, H, W, Cout, Cout, 0);
    };
    auto run_pool = [&](const half_t* src, half_t* dst, int Hs, int Ws, int C) {
        int tot = 4 * (Hs / 2) * (Ws / 2) * C;
        pool_kernel<<<(tot + 255) / 256, 256, 0, stream>>>(src, dst, Hs, Ws, C, tot);
    };
    auto run_up = [&](const half_t* src, half_t* dst, const float* tb,
                      int Hs, int Ws, int Cs, int Cdst) {
        int tot = 4 * (2 * Hs) * (2 * Ws) * Cs;
        up2_kernel<<<(tot + 255) / 256, 256, 0, stream>>>(src, dst, tb, Hs, Ws, Cs, Cdst, tot);
    };
    auto run_cat = [&](const half_t* src, half_t* dst, int HWtot, int Csrc, int Cdst, int coff) {
        int tot = HWtot * Csrc;
        concat_copy_kernel<<<(tot + 255) / 256, 256, 0, stream>>>(src, dst, Csrc, Cdst, coff, tot);
    };

    // ---------------- UNet ----------------
    run_block(xin, 0, 7, 256, 256, 64, e1b);
    run_pool(e1b, p1b, 256, 256, 64);
    run_block(p1b, 2, 15, 128, 128, 128, e2b);
    run_pool(e2b, p2b, 128, 128, 128);
    run_block(p2b, 4, 23, 64, 64, 256, e3b);
    run_pool(e3b, p3b, 64, 64, 256);
    run_block(p3b, 6, 31, 32, 32, 256, botb);

    run_up(botb, d3in, temb, 32, 32, 256, 512);          // up2(bott + temb)
    run_cat(e3b, d3in, 4 * 4096, 256, 512, 256);
    run_block(d3in, 8, 39, 64, 64, 128, d3out);

    run_up(d3out, d2in, nullptr, 64, 64, 128, 256);
    run_cat(e2b, d2in, 4 * 16384, 128, 256, 128);
    run_block(d2in, 10, 47, 128, 128, 64, d2out);

    run_up(d2out, d1in, nullptr, 128, 128, 64, 128);
    run_cat(e1b, d1in, 4 * 65536, 64, 128, 64);
    run_block(d1in, 12, 55, 256, 256, 64, d1out);

    {   // 1x1 output conv
        int tot = 4 * 65536 * 2;
        out_conv_kernel<<<(tot + 255) / 256, 256, 0, stream>>>(
            d1out, (const float*)d_in[63], (const float*)d_in[64], npred, 65536, tot);
    }

    // ---------------- diffusion tail ----------------
    coef_kernel<<<1, 32, 0, stream>>>(tarr, cf);
    {
        int tot = 4 * 65536;
        x0_kernel<<<(tot + 255) / 256, 256, 0, stream>>>(x_t, npred, cf, vx, vy, 65536, tot);
        div_kernel<<<(tot + 255) / 256, 256, 0, stream>>>(vx, vy, dvb, 256, 256, tot);
        jacobi_init_kernel<<<(tot + 255) / 256, 256, 0, stream>>>(dvb, phiA, tot);
        float* pa = phiA; float* pb = phiB;
        for (int it = 1; it < 50; ++it) {
            jacobi_kernel<<<(tot + 255) / 256, 256, 0, stream>>>(pa, dvb, pb, 256, 256, tot);
            float* tmp = pa; pa = pb; pb = tmp;
        }
        final_kernel<<<(tot + 255) / 256, 256, 0, stream>>>(pa, vx, vy, npred, noise, cf,
                                                            out, 256, 256, tot);
    }
}
